// CausalSelfAttention_53884659696251
// MI455X (gfx1250) — matrix operand order repaired and checked by reading, not compiled, not run on hardware
//
#include <hip/hip_runtime.h>
#include <hip/hip_bf16.h>
#include <math.h>

typedef __attribute__((ext_vector_type(16))) __bf16 bf16x16;
typedef __attribute__((ext_vector_type(8)))  float  f32x8;

static constexpr int Bsz = 4, Tseq = 2048, Cdim = 1024, Hn = 16, Dh = 64;
static constexpr int Mtot = Bsz * Tseq;   // 8192
static constexpr int N1   = 3 * Cdim;     // 3072

__device__ __forceinline__ unsigned short f2bf(float f) {
  union { float f; unsigned u; } v; v.f = f;
  unsigned u = v.u;
  u += 0x7fffu + ((u >> 16) & 1u);        // round-to-nearest-even
  return (unsigned short)(u >> 16);
}

union FragBf { bf16x16 v; uint4 q[2]; unsigned short s[16]; };

// A-matrix 16x32 fragment (ISA 7.12.2): lane row = lane&15,
// K(e) = half*8 + (e<8 ? e : e+8)  -> two contiguous 16B chunks.
__device__ __forceinline__ bf16x16 make_fragA(const unsigned short* row, int half) {
  FragBf f;
  f.q[0] = *(const uint4*)(row + half * 8);
  f.q[1] = *(const uint4*)(row + half * 8 + 16);
  return f.v;
}
// B-matrix 32x16 fragment: lane col = lane&15, K(e) = half*8 + (e<8 ? e : e+8)
// (row ptr points at [col][k0] of a K-contiguous, N-major (transposed) tile)
__device__ __forceinline__ bf16x16 make_fragB(const unsigned short* row, int half) {
  FragBf f;
  f.q[0] = *(const uint4*)(row + half * 8);
  f.q[1] = *(const uint4*)(row + half * 8 + 16);
  return f.v;
}

// Async global->LDS copy of 16 bytes (CDNA5 GLOBAL_LOAD_ASYNC_TO_LDS_B128,
// tracked by ASYNCcnt). lds_off = 32-bit LDS byte address (low 32 bits of the
// generic pointer), gaddr = 64-bit global address (GV mode, saddr = off).
__device__ __forceinline__ void async_copy_b128(void* lds_ptr, const void* gaddr) {
  unsigned lds_off = (unsigned)(uintptr_t)lds_ptr;
  asm volatile("global_load_async_to_lds_b128 %0, %1, off"
               :: "v"(lds_off), "v"(gaddr) : "memory");
}

// ---------------------------------------------------------------------------
// One-time conversion kernels (run once; remove all f2bf from GEMM hot loops)
// ---------------------------------------------------------------------------
__global__ __launch_bounds__(256)
void cvt_bf16_kernel(const float* __restrict__ in, unsigned short* __restrict__ out,
                     int n4) {
  int i = blockIdx.x * 256 + threadIdx.x;
  if (i >= n4) return;
  float4 f = ((const float4*)in)[i];
  unsigned short o[4] = { f2bf(f.x), f2bf(f.y), f2bf(f.z), f2bf(f.w) };
  ((uint2*)out)[i] = *(const uint2*)o;
}

// out[n][k] = bf16(in[k][n])   (in: K x N f32, out: N x K bf16)
__global__ __launch_bounds__(256)
void cvt_t_bf16_kernel(const float* __restrict__ in, unsigned short* __restrict__ out,
                       int K, int N) {
  int n4cnt = N >> 2;
  int gid = blockIdx.x * 256 + threadIdx.x;
  if (gid >= K * n4cnt) return;
  int k = gid / n4cnt;
  int n = (gid - k * n4cnt) * 4;
  float4 f = *(const float4*)(in + (size_t)k * N + n);
  out[(size_t)(n + 0) * K + k] = f2bf(f.x);
  out[(size_t)(n + 1) * K + k] = f2bf(f.y);
  out[(size_t)(n + 2) * K + k] = f2bf(f.z);
  out[(size_t)(n + 3) * K + k] = f2bf(f.w);
}

// ---------------------------------------------------------------------------
// Shared 128x128 GEMM core: async double-buffered LDS, bf16 WMMA, K-step 32.
// Ag: 128 rows [m][Cdim] bf16; Bg: 128 rows [n][Cdim] bf16 (pre-transposed W).
// ---------------------------------------------------------------------------
__device__ __forceinline__ void gemm_core_128x128(
    const unsigned short* __restrict__ Ag, const unsigned short* __restrict__ Bg,
    unsigned short (*As)[128][32], unsigned short (*Bs)[128][32],
    int tid, f32x8 acc[4][4]) {
  const int lane = tid & 31;
  const int wid  = tid >> 5;
  const int half = lane >> 4;
  const int ln   = lane & 15;
  const int wm   = (wid >> 1) * 64;
  const int wn   = (wid & 1) * 64;

  // issue one K-step's tiles (8 async b128 per thread)
  auto issue = [&](int s, int buf) {
    const int k0 = s * 32;
    for (int i = 0; i < 4; i++) {
      int idx = tid + i * 128;
      int row = idx >> 2;
      int c8  = (idx & 3) * 8;
      async_copy_b128(&As[buf][row][c8], Ag + (size_t)row * Cdim + k0 + c8);
      async_copy_b128(&Bs[buf][row][c8], Bg + (size_t)row * Cdim + k0 + c8);
    }
  };

  const int nsteps = Cdim / 32;
  issue(0, 0);
  for (int s = 0; s < nsteps; s++) {
    const int cur = s & 1;
    if (s + 1 < nsteps) {
      issue(s + 1, cur ^ 1);                       // prefetch next buffer
      asm volatile("s_wait_asynccnt 8" ::: "memory");  // current buffer landed
    } else {
      asm volatile("s_wait_asynccnt 0" ::: "memory");
    }
    __syncthreads();

    bf16x16 afr[4];
    for (int mi = 0; mi < 4; mi++)
      afr[mi] = make_fragA(&As[cur][wm + mi * 16 + ln][0], half);
    for (int ni = 0; ni < 4; ni++) {
      bf16x16 bfr = make_fragB(&Bs[cur][wn + ni * 16 + ln][0], half);
      for (int mi = 0; mi < 4; mi++)
        acc[mi][ni] = __builtin_amdgcn_wmma_f32_16x16x32_bf16(
            false, afr[mi], false, bfr, (short)0, acc[mi][ni], false, false);
    }
    __syncthreads();   // done reading 'cur' before it is overwritten
  }
}

// ---------------------------------------------------------------------------
// Kernel 1: qkv = x @ w_attn + b_attn, scattered into Q/K/V [B,H,T,D] bf16
// ---------------------------------------------------------------------------
__global__ __launch_bounds__(128)
void qkv_gemm_kernel(const unsigned short* __restrict__ xb,
                     const unsigned short* __restrict__ wat,  // [N1][Cdim] bf16
                     const float* __restrict__ bias,
                     unsigned short* __restrict__ Qb,
                     unsigned short* __restrict__ Kb,
                     unsigned short* __restrict__ Vb) {
  __shared__ alignas(16) unsigned short As[2][128][32];
  __shared__ alignas(16) unsigned short Bs[2][128][32];

  const int tid  = threadIdx.x;
  const int lane = tid & 31;
  const int wid  = tid >> 5;
  const int half = lane >> 4;
  const int ln   = lane & 15;
  const int n0   = blockIdx.x * 128;
  const int m0   = blockIdx.y * 128;
  const int wm   = (wid >> 1) * 64;
  const int wn   = (wid & 1) * 64;

  f32x8 acc[4][4] = {};
  gemm_core_128x128(xb + (size_t)m0 * Cdim, wat + (size_t)n0 * Cdim,
                    As, Bs, tid, acc);

  for (int mi = 0; mi < 4; mi++)
    for (int ni = 0; ni < 4; ni++)
      for (int r = 0; r < 8; r++) {
        int mg = m0 + wm + mi * 16 + r + half * 8;
        int ng = n0 + wn + ni * 16 + ln;
        float val = acc[mi][ni][r] + bias[ng];
        int sel = ng >> 10;           // 0:q 1:k 2:v
        int ci  = ng & 1023;
        int hh  = ci >> 6, dd = ci & 63;
        int bb  = mg >> 11, tt = mg & 2047;   // T = 2048
        unsigned short* dst = (sel == 0) ? Qb : (sel == 1) ? Kb : Vb;
        dst[((size_t)(bb * Hn + hh) * Tseq + tt) * Dh + dd] = f2bf(val);
      }
}

// ---------------------------------------------------------------------------
// Kernel 2: flash attention, causal, per (b, h, 64 q-rows); 16 q-rows / wave
// ---------------------------------------------------------------------------
__global__ __launch_bounds__(128)
void flash_attn_kernel(const unsigned short* __restrict__ Qb,
                       const unsigned short* __restrict__ Kb,
                       const unsigned short* __restrict__ Vb,
                       unsigned short* __restrict__ Yb) {
  __shared__ alignas(16) unsigned short Kt[64][64];      // [key][d]
  __shared__ alignas(16) unsigned short Vt[64][64];      // [d][key] (transposed)
  __shared__ alignas(16) unsigned short Pl[4][16][64];   // per-wave P scratch

  const int tid = threadIdx.x, lane = tid & 31, wid = tid >> 5;
  const int half = lane >> 4, ln = lane & 15;
  const int qb = blockIdx.x, h = blockIdx.y, b = blockIdx.z;
  const size_t headOff = ((size_t)(b * Hn + h)) * Tseq * Dh;
  const int q0 = qb * 64 + wid * 16;

  // Q held in registers for the whole pass (A-fragment layout, Kdim = D = 2x32)
  bf16x16 qa[2];
  {
    const unsigned short* qrow = Qb + headOff + (size_t)(q0 + ln) * Dh;
    qa[0] = make_fragA(qrow, half);
    qa[1] = make_fragA(qrow + 32, half);
  }

  f32x8 ofr[4] = {};
  float mrow[8], lrow[8];
  for (int r = 0; r < 8; r++) { mrow[r] = -3.0e38f; lrow[r] = 0.f; }

  const float sc = 0.125f * 1.44269504088896340736f;  // 1/sqrt(64) * log2(e)

  for (int j = 0; j <= qb; j++) {
    const int k0 = j * 64;
    // K tile: straight copy -> async global->LDS. V tile: transposed manually.
    for (int i = 0; i < 4; i++) {
      int idx = tid + i * 128;           // 512 x 16B
      int key = idx >> 3;
      int c8  = (idx & 7) * 8;
      async_copy_b128(&Kt[key][c8], Kb + headOff + (size_t)(k0 + key) * Dh + c8);
      uint4 vv = *(const uint4*)(Vb + headOff + (size_t)(k0 + key) * Dh + c8);
      const unsigned short* vs = (const unsigned short*)&vv;
      for (int e = 0; e < 8; e++) Vt[c8 + e][key] = vs[e];
    }
    asm volatile("s_wait_asynccnt 0" ::: "memory");
    __syncthreads();

    // S = Q @ K^T  (B-fragment reads Kt[key][d] contiguously in d)
    f32x8 sfr[4] = {};
    for (int ni = 0; ni < 4; ni++)
      for (int c = 0; c < 2; c++) {
        bf16x16 bk = make_fragB(&Kt[ni * 16 + ln][c * 32], half);
        sfr[ni] = __builtin_amdgcn_wmma_f32_16x16x32_bf16(
            false, qa[c], false, bk, (short)0, sfr[ni], false, false);
      }

    // scale + causal mask + row max (reduction stays in 16-lane half)
    float newm[8];
    for (int r = 0; r < 8; r++) {
      int qg = q0 + r + half * 8;
      float mx = -3.0e38f;
      for (int ni = 0; ni < 4; ni++) {
        int kg = k0 + ni * 16 + ln;
        float s = sfr[ni][r] * sc;
        if (kg > qg) s = -3.0e38f;
        sfr[ni][r] = s;
        mx = fmaxf(mx, s);
      }
      for (int off = 1; off < 16; off <<= 1)
        mx = fmaxf(mx, __shfl_xor(mx, off, 32));
      newm[r] = fmaxf(mrow[r], mx);
    }

    // P = 2^(s - m), row sums, O rescale, P -> per-wave LDS (bf16)
    for (int r = 0; r < 8; r++) {
      float corr = exp2f(mrow[r] - newm[r]);
      float rsum = 0.f;
      for (int ni = 0; ni < 4; ni++) {
        float p = exp2f(sfr[ni][r] - newm[r]);
        rsum += p;
        Pl[wid][r + half * 8][ni * 16 + ln] = f2bf(p);
      }
      for (int off = 1; off < 16; off <<= 1)
        rsum += __shfl_xor(rsum, off, 32);
      lrow[r] = lrow[r] * corr + rsum;
      mrow[r] = newm[r];
      for (int ni = 0; ni < 4; ni++) ofr[ni][r] = ofr[ni][r] * corr;
    }
    // same-wave LDS transpose: wait own ds stores, no block barrier needed
    asm volatile("s_wait_dscnt 0" ::: "memory");

    // O += P @ V   (Vt rows contiguous in key)
    for (int c = 0; c < 2; c++) {
      bf16x16 pa = make_fragA(&Pl[wid][ln][c * 32], half);
      for (int ni = 0; ni < 4; ni++) {
        bf16x16 bv = make_fragB(&Vt[ni * 16 + ln][c * 32], half);
        ofr[ni] = __builtin_amdgcn_wmma_f32_16x16x32_bf16(
            false, pa, false, bv, (short)0, ofr[ni], false, false);
      }
    }
    __syncthreads();
  }

  // normalize and store y [B,T,C] bf16 (heads interleaved back to C)
  for (int ni = 0; ni < 4; ni++)
    for (int r = 0; r < 8; r++) {
      int M = r + half * 8;
      float v = ofr[ni][r] / lrow[r];
      int row = q0 + M;
      int col = h * Dh + ni * 16 + ln;
      Yb[((size_t)(b * Tseq) + row) * Cdim + col] = f2bf(v);
    }
}

// ---------------------------------------------------------------------------
// Kernel 3: out = y @ w_proj + b_proj (f32 output)
// ---------------------------------------------------------------------------
__global__ __launch_bounds__(128)
void proj_gemm_kernel(const unsigned short* __restrict__ Yb,
                      const unsigned short* __restrict__ wpt,  // [Cdim][Cdim] bf16
                      const float* __restrict__ bias,
                      float* __restrict__ out) {
  __shared__ alignas(16) unsigned short As[2][128][32];
  __shared__ alignas(16) unsigned short Bs[2][128][32];

  const int tid  = threadIdx.x;
  const int lane = tid & 31;
  const int wid  = tid >> 5;
  const int half = lane >> 4;
  const int ln   = lane & 15;
  const int n0   = blockIdx.x * 128;
  const int m0   = blockIdx.y * 128;
  const int wm   = (wid >> 1) * 64;
  const int wn   = (wid & 1) * 64;

  f32x8 acc[4][4] = {};
  gemm_core_128x128(Yb + (size_t)m0 * Cdim, wpt + (size_t)n0 * Cdim,
                    As, Bs, tid, acc);

  for (int mi = 0; mi < 4; mi++)
    for (int ni = 0; ni < 4; ni++)
      for (int r = 0; r < 8; r++) {
        int mg = m0 + wm + mi * 16 + r + half * 8;
        int ng = n0 + wn + ni * 16 + ln;
        out[(size_t)mg * Cdim + ng] = acc[mi][ni][r] + bias[ng];
      }
}

// ---------------------------------------------------------------------------
extern "C" void kernel_launch(void* const* d_in, const int* in_sizes, int n_in,
                              void* d_out, int out_size, void* d_ws, size_t ws_size,
                              hipStream_t stream) {
  (void)in_sizes; (void)n_in; (void)out_size; (void)ws_size;
  const float* x      = (const float*)d_in[0];
  const float* w_attn = (const float*)d_in[1];
  const float* b_attn = (const float*)d_in[2];
  const float* w_proj = (const float*)d_in[3];
  const float* b_proj = (const float*)d_in[4];
  float* out = (float*)d_out;

  unsigned short* ws = (unsigned short*)d_ws;
  const size_t nXB = (size_t)Mtot * Cdim;        // 8,388,608
  const size_t nWA = (size_t)N1 * Cdim;          // 3,145,728
  const size_t nWP = (size_t)Cdim * Cdim;        // 1,048,576
  unsigned short* xb  = ws;
  unsigned short* wat = xb  + nXB;               // w_attn transposed [N1][C]
  unsigned short* wpt = wat + nWA;               // w_proj transposed [C][C]
  unsigned short* Qb  = wpt + nWP;
  unsigned short* Kb  = Qb + nXB;
  unsigned short* Vb  = Kb + nXB;
  unsigned short* Yb  = Vb + nXB;

  // one-time bf16 conversions (weights pre-transposed to [n][k])
  cvt_bf16_kernel<<<(int)(nXB / 4 / 256), 256, 0, stream>>>(x, xb, (int)(nXB / 4));
  cvt_t_bf16_kernel<<<(int)(nWA / 4 / 256), 256, 0, stream>>>(w_attn, wat, Cdim, N1);
  cvt_t_bf16_kernel<<<(int)(nWP / 4 / 256), 256, 0, stream>>>(w_proj, wpt, Cdim, Cdim);

  qkv_gemm_kernel<<<dim3(N1 / 128, Mtot / 128), 128, 0, stream>>>(
      xb, wat, b_attn, Qb, Kb, Vb);
  flash_attn_kernel<<<dim3(Tseq / 64, Hn, Bsz), 128, 0, stream>>>(
      Qb, Kb, Vb, Yb);
  proj_gemm_kernel<<<dim3(Cdim / 128, Mtot / 128), 128, 0, stream>>>(
      Yb, w_proj ? wpt : wpt, b_proj, out);
}
